// RNN_41781441855808
// MI455X (gfx1250) — matrix operand order repaired and checked by reading, not compiled, not run on hardware
//
#include <hip/hip_runtime.h>

// Problem dims (match reference)
constexpr int BATCH = 64;
constexpr int SEQ   = 2048;
constexpr int KIN   = 256;   // input feature dim
constexpr int OUT   = 256;   // hidden/output dim

typedef __attribute__((ext_vector_type(16))) __bf16 v16bf;
typedef __attribute__((ext_vector_type(8)))  __bf16 v8bf;
typedef __attribute__((ext_vector_type(8)))  float  v8f;

// gfx1250 async global->LDS copy (ASYNCcnt-tracked), GV addressing mode:
// first operand = VGPR holding LDS byte address, second = 64-bit global address.
#define ASYNC_LOAD_B128(ldsaddr_u32, gaddr_ptr)                                  \
    asm volatile("global_load_async_to_lds_b128 %0, %1, off"                     \
                 :: "v"(ldsaddr_u32), "v"(gaddr_ptr) : "memory")
#define WAIT_ASYNCCNT0() asm volatile("s_wait_asynccnt 0" ::: "memory")

// CDNA5 split workgroup barrier: announce arrival once our LDS ops are done,
// do independent work, then wait for the rest of the workgroup.
#define WG_SIGNAL_AFTER_DS() \
    asm volatile("s_wait_dscnt 0x0\n\ts_barrier_signal -1" ::: "memory")
#define WG_WAIT() asm volatile("s_barrier_wait -1" ::: "memory")

// Low 32 bits of a generic pointer to __shared__ == LDS byte address (AS3 offset).
__device__ __forceinline__ unsigned lds_addr(const void* p) {
    return (unsigned)(unsigned long long)p;
}

__device__ __forceinline__ float tanh_fast(float x) {
#if __has_builtin(__builtin_amdgcn_tanhf)
    return __builtin_amdgcn_tanhf(x);     // V_TANH_F32 on gfx1250
#else
    return tanhf(x);
#endif
}

// Assemble one 16x32 bf16 A-chunk per the CDNA5 16-bit A layout:
// lane = 16*kh + m ; VGPR0-3 hold K = c*32 + kh*8 + (0..7), VGPR4-7 hold +16.
__device__ __forceinline__ v16bf load_a_chunk(const __bf16* arow, int c, int kh) {
    const v8bf lo = *(const v8bf*)(arow + c * 32 + kh * 8);
    const v8bf hi = *(const v8bf*)(arow + c * 32 + 16 + kh * 8);
    return __builtin_shufflevector(lo, hi, 0, 1, 2, 3, 4, 5, 6, 7,
                                           8, 9, 10, 11, 12, 13, 14, 15);
}

// Preload a 256x16 bf16 B slice (columns [16w, 16w+16)) into 8 register chunks.
// B layout (32x16 bf16): lane = 16*kh + n ; elements 0-7 hold K = kh*8 + (0..7), 8-15 hold +16.
__device__ __forceinline__ void load_b_regs(const float* __restrict__ W, int ncol,
                                            int kh, v16bf breg[8]) {
#pragma unroll
    for (int c = 0; c < 8; ++c) {
        const int kb = c * 32 + kh * 8;
        v16bf b;
#pragma unroll
        for (int j = 0; j < 16; ++j)
            b[j] = (__bf16)W[(kb + j + (j >= 8 ? 8 : 0)) * OUT + ncol];
        breg[c] = b;
    }
}

// ---------------- Phase 1: zx = input @ W_x + (b_x + b_h), written into d_out[y] ----------
// A-tile double-buffered in LDS: next tile's global loads overlap current tile's WMMAs;
// one barrier per tile.
__global__ __launch_bounds__(512) void rnn_zx_gemm(const float* __restrict__ in,
                                                   const float* __restrict__ Wx,
                                                   const float* __restrict__ bh,
                                                   const float* __restrict__ bx,
                                                   float* __restrict__ out) {
    __shared__ __align__(16) __bf16 a_lds[2][16 * KIN];   // 2 x 8KB

    const int tid  = threadIdx.x;
    const int wave = tid >> 5;
    const int lane = tid & 31;
    const int nh   = lane >> 4;      // half selector (K-half for A/B, M-half for C)
    const int nl   = lane & 15;
    const int ncol = wave * 16 + nl;

    v16bf breg[8];
    load_b_regs(Wx, ncol, nh, breg);
    const float bias = bh[ncol] + bx[ncol];

    const int mtiles = (BATCH * SEQ) / 16;   // 8192
    const int stride = (int)gridDim.x;
    const int e0 = tid * 8;                  // staging: 8 consecutive bf16 per thread
    const int sm = e0 >> 8, sk = e0 & 255;

    // Prologue: stage the first tile into buffer 0.
    {
        const long rowbase = (long)blockIdx.x * 16;
        v8bf pk;
#pragma unroll
        for (int j = 0; j < 8; ++j)
            pk[j] = (__bf16)in[(rowbase + sm) * KIN + sk + j];
        *(v8bf*)&a_lds[0][e0] = pk;
    }
    __syncthreads();

    int cb = 0;
    for (int mt = blockIdx.x; mt < mtiles; mt += stride, cb ^= 1) {
        const long rowbase  = (long)mt * 16;
        const int  nxt      = mt + stride;
        const bool havenext = nxt < mtiles;   // uniform across WG

        // Issue next tile's global loads now; they complete under the WMMAs.
        float sreg[8];
        if (havenext) {
#pragma unroll
            for (int j = 0; j < 8; ++j)
                sreg[j] = in[((long)nxt * 16 + sm) * KIN + sk + j];
        }

        // Compute on current buffer.
        const __bf16* arow = &a_lds[cb][nl * KIN];
        v16bf a[8];
#pragma unroll
        for (int c = 0; c < 8; ++c)
            a[c] = load_a_chunk(arow, c, nh);

        v8f acc;
#pragma unroll
        for (int r = 0; r < 8; ++r) acc[r] = bias;
#pragma unroll
        for (int c = 0; c < 8; ++c)
            acc = __builtin_amdgcn_wmma_f32_16x16x32_bf16(false, a[c], false, breg[c],
                                                          (short)0, acc, false, false);
#pragma unroll
        for (int r = 0; r < 8; ++r) {
            const int m = r + nh * 8;        // C layout: lanes 16-31 are M+8
            out[(rowbase + m) * OUT + ncol] = acc[r];
        }

        if (havenext) {
            // Stage next tile into the other buffer (no reader conflict this iteration).
            v8bf pk;
#pragma unroll
            for (int j = 0; j < 8; ++j)
                pk[j] = (__bf16)sreg[j];
            *(v8bf*)&a_lds[cb ^ 1][e0] = pk;
            WG_SIGNAL_AFTER_DS();   // my reads of cb + writes of nb retired
            WG_WAIT();              // everyone's writes of nb visible
        }
    }
}

// ---------------- Phase 2: sequential scan; h double-buffered in LDS, W_h in registers,
// zx double-buffered in LDS via async global->LDS loads. One split barrier per step. -------
__global__ __launch_bounds__(512) void rnn_scan(float* __restrict__ yzx,
                                                float* __restrict__ hfin,
                                                const float* __restrict__ Wh,
                                                const float* __restrict__ h0) {
    __shared__ __align__(16) __bf16 h_sh[2][16 * OUT];    // 2 x 8KB
    __shared__ __align__(16) float  zx_sh[2][16 * OUT];   // 2 x 16KB

    const int tid  = threadIdx.x;
    const int wave = tid >> 5;
    const int lane = tid & 31;
    const int nh   = lane >> 4;
    const int nl   = lane & 15;
    const int ncol = wave * 16 + nl;
    const int rowbase = blockIdx.x * 16;

    v16bf breg[8];
    load_b_regs(Wh, ncol, nh, breg);

    // Stage h0 (bf16) into h buffer 0.
#pragma unroll
    for (int j = 0; j < 8; ++j) {
        const int e = tid * 8 + j;
        h_sh[0][e] = (__bf16)h0[(rowbase + (e >> 8)) * OUT + (e & 255)];
    }

    // Async zx staging: thread covers 16B at byte tid*16 of the 16KB tile (rows 0..7)
    // and the mirror 16B at +8KB (rows 8..15). Row stride in global = SEQ*OUT floats.
    const int   am = tid >> 6;                 // tile row 0..7 for first half
    const int   ac = (tid & 63) * 4;           // float column 0..252
    const float* gA = yzx + (size_t)(rowbase + am) * SEQ * OUT + ac;
    const float* gB = yzx + (size_t)(rowbase + am + 8) * SEQ * OUT + ac;
    const unsigned ldsA0 = lds_addr(&zx_sh[0][tid * 4]);
    const unsigned ldsB0 = lds_addr(&zx_sh[0][8 * OUT + tid * 4]);
    const unsigned ldsA1 = lds_addr(&zx_sh[1][tid * 4]);
    const unsigned ldsB1 = lds_addr(&zx_sh[1][8 * OUT + tid * 4]);

    // Prime buffer 0 with zx_{t=0}.
    ASYNC_LOAD_B128(ldsA0, gA);
    ASYNC_LOAD_B128(ldsB0, gB);
    WAIT_ASYNCCNT0();
    __syncthreads();   // h0 staged + zx_0 resident

    float res[8];
#pragma unroll
    for (int r = 0; r < 8; ++r) res[r] = 0.0f;

    for (int t = 0; t < SEQ; ++t) {
        const int cb = t & 1;
        const int nb = cb ^ 1;

        // Kick off zx_{t+1} into the other buffer; latency hidden behind this step.
        if (t + 1 < SEQ) {                                   // uniform branch
            const float* g1 = gA + (size_t)(t + 1) * OUT;
            const float* g2 = gB + (size_t)(t + 1) * OUT;
            ASYNC_LOAD_B128(cb ? ldsA0 : ldsA1, g1);
            ASYNC_LOAD_B128(cb ? ldsB0 : ldsB1, g2);
        }

        // acc starts as zx_t (biases already folded in), read from LDS.
        v8f acc;
#pragma unroll
        for (int r = 0; r < 8; ++r)
            acc[r] = zx_sh[cb][(r + nh * 8) * OUT + ncol];

        // h_{t-1} @ W_h over K=256: 8 chained WMMAs on the current h buffer.
        const __bf16* arow = &h_sh[cb][nl * OUT];
        v16bf a[8];
#pragma unroll
        for (int c = 0; c < 8; ++c)
            a[c] = load_a_chunk(arow, c, nh);
#pragma unroll
        for (int c = 0; c < 8; ++c)
            acc = __builtin_amdgcn_wmma_f32_16x16x32_bf16(false, a[c], false, breg[c],
                                                          (short)0, acc, false, false);

        // New h goes to the other buffer: no read/write hazard, no mid-step barrier.
#pragma unroll
        for (int r = 0; r < 8; ++r) {
            const int m = r + nh * 8;
            const float v = tanh_fast(acc[r]);
            res[r] = v;
            h_sh[nb][m * OUT + ncol] = (__bf16)v;
        }

        WG_SIGNAL_AFTER_DS();   // my LDS reads (zx_t, h_t-1) + writes (h_t) retired

        // Overlap with barrier propagation: fire-and-forget y stores + async wait.
#pragma unroll
        for (int r = 0; r < 8; ++r) {
            const int m = r + nh * 8;
            yzx[(long)(rowbase + m) * SEQ * OUT + (long)t * OUT + ncol] = res[r];
        }
        WAIT_ASYNCCNT0();       // my zx_{t+1} portion has landed in LDS

        WG_WAIT();              // everyone's h_t + zx_{t+1} ready
    }

    // h_final = last y
#pragma unroll
    for (int r = 0; r < 8; ++r) {
        const int m = r + nh * 8;
        hfin[(rowbase + m) * OUT + ncol] = res[r];
    }
}

extern "C" void kernel_launch(void* const* d_in, const int* in_sizes, int n_in,
                              void* d_out, int out_size, void* d_ws, size_t ws_size,
                              hipStream_t stream) {
    const float* input = (const float*)d_in[0];  // [64,2048,256]
    const float* h0    = (const float*)d_in[1];  // [64,256]
    const float* Wh    = (const float*)d_in[2];  // [256,256]
    const float* Wx    = (const float*)d_in[3];  // [256,256]
    const float* bh    = (const float*)d_in[4];  // [256]
    const float* bx    = (const float*)d_in[5];  // [256]
    float* out = (float*)d_out;                  // y [64,2048,256] then h_final [64,256]

    // Phase 1: zx into the y region of d_out (consumed & overwritten in phase 2).
    rnn_zx_gemm<<<256, 512, 0, stream>>>(input, Wx, bh, bx, out);
    // Phase 2: 4 WGs, one per 16 independent batch-row chains.
    rnn_scan<<<BATCH / 16, 512, 0, stream>>>(out, out + (size_t)BATCH * SEQ * OUT, Wh, h0);
}
